// PointNet2Encoder_50010599195040
// MI455X (gfx1250) — matrix operand order repaired and checked by reading, not compiled, not run on hardware
//
#include <hip/hip_runtime.h>
#include <hip/hip_bf16.h>

// ---------------------------------------------------------------------------
// PointNet++ encoder for MI455X (gfx1250, wave32).
// All GEMMs (SA shared-MLPs + FC head) run on v_wmma_f32_16x16x32_f16 with
// BN folded into f16 weights pre-swizzled into WMMA B-fragment layout.
// A-fragments are loaded from LDS once per (mt,kt) and reused across all
// N-tiles owned by the wave (register-resident accumulators). Wave index is
// readfirstlane'd so all per-N-tile guards are scalar branches (EXEC stays
// all-ones at every WMMA site, no accumulator copy bloat).
// ---------------------------------------------------------------------------

typedef __attribute__((ext_vector_type(16))) _Float16 v16h;
typedef __attribute__((ext_vector_type(8)))  float    v8f;

#define EPSF 1e-5f

__device__ __forceinline__ v8f wmma_f32_16x16x32_f16(v16h a, v16h b, v8f c) {
  // 8 args: (neg_a, A, neg_b, B, c_mod, C, reuse_a, reuse_b)
  return __builtin_amdgcn_wmma_f32_16x16x32_f16(false, a, false, b, (short)0, c,
                                                false, false);
}

__device__ __forceinline__ v8f vzero8() {
  v8f z = {0.f, 0.f, 0.f, 0.f, 0.f, 0.f, 0.f, 0.f};
  return z;
}

// Lane-uniform wave index in SGPR: guards using it become scalar branches.
__device__ __forceinline__ int wave_id(int tid) {
  return __builtin_amdgcn_readfirstlane(tid >> 5);
}

// A-fragment (16x32 f16, M x K) from row-major LDS tile act[row*ldk + k].
// ISA layout: lanes 0-15 rows M=0..15 with halfs 0..7 = K(base..base+7),
// halfs 8..15 = K(16+base..16+base+7); base = 8 for lanes 16..31.
__device__ __forceinline__ v16h load_A_frag(const _Float16* lds, int ldk,
                                            int mt, int kt, int lane) {
  const int row = mt * 16 + (lane & 15);
  const int kb  = kt * 32 + ((lane >> 4) << 3);
  union { v16h v; unsigned u[8]; } r;
  const unsigned* p0 = (const unsigned*)(lds + row * ldk + kb);
  const unsigned* p1 = (const unsigned*)(lds + row * ldk + kb + 16);
#pragma unroll
  for (int i = 0; i < 4; ++i) { r.u[i] = p0[i]; r.u[4 + i] = p1[i]; }
  return r.v;
}

// ---------------------------------------------------------------------------
// Weight prep: fold BN into f16 weights in WMMA B-fragment order.
// Fragment layout: element((kt*NT+nt)*32 + lane)*16 + h  holds
//   B[k][n] with n = nt*16 + lane%16, k = kt*32 + (lane>=16?8:0) + (h&7) + (h>=8?16:0)
// (same K order as the A-fragment: halfs 0..7 = K(base..base+7), 8..15 = K(16+base..)).
// ---------------------------------------------------------------------------
__global__ void fold_weights_kernel(const float* __restrict__ W,
                                    const float* __restrict__ gamma,
                                    const float* __restrict__ beta,
                                    const float* __restrict__ mean,
                                    const float* __restrict__ var,
                                    const float* __restrict__ lin_b,
                                    _Float16* __restrict__ Wfrag,
                                    float* __restrict__ bias,
                                    int cin, int kpad, int cout) {
  const int i = blockIdx.x * blockDim.x + threadIdx.x;
  const int total = kpad * cout;
  if (i < total) {
    const int NT = cout >> 4;
    int e = i;
    const int h    = e & 15;  e >>= 4;
    const int lane = e & 31;  e >>= 5;
    const int nt   = e % NT;
    const int kt   = e / NT;
    const int n = nt * 16 + (lane & 15);
    const int k = kt * 32 + ((lane >> 4) << 3) + (h & 7) + ((h >> 3) << 4);
    float v = 0.f;
    if (k < cin) {
      float scale = 1.f;
      if (gamma) scale = gamma[n] * rsqrtf(var[n] + EPSF);
      v = W[(size_t)n * cin + k] * scale;
    }
    Wfrag[i] = (_Float16)v;
  }
  if (i < cout) {
    float bv;
    if (gamma) {
      const float scale = gamma[i] * rsqrtf(var[i] + EPSF);
      bv = beta[i] - mean[i] * scale;
    } else {
      bv = lin_b[i];
    }
    bias[i] = bv;
  }
}

// ---------------------------------------------------------------------------
// Farthest point sampling: one block per batch, sequential over npoint steps.
// Matches reference semantics (index 0 emitted first).
// ---------------------------------------------------------------------------
template <int N, int NPOINT>
__global__ void __launch_bounds__(256) fps_kernel(const float* __restrict__ xyz,
                                                  float* __restrict__ new_xyz) {
  constexpr int T = 256;
  constexpr int PPT = N / T;
  const int b = blockIdx.x;
  const int t = threadIdx.x;
  const float* p = xyz + (size_t)b * N * 3;

  float px[PPT], py[PPT], pz[PPT], dists[PPT];
#pragma unroll
  for (int i = 0; i < PPT; ++i) {
    const int pi = t + i * T;
    px[i] = p[pi * 3 + 0];
    py[i] = p[pi * 3 + 1];
    pz[i] = p[pi * 3 + 2];
    dists[i] = 1e10f;
  }

  __shared__ float redv[T];
  __shared__ int   redi[T];
  __shared__ int   s_last;
  if (t == 0) s_last = 0;

  for (int s = 0; s < NPOINT; ++s) {
    __syncthreads();
    const int last = s_last;
    const float cx = p[last * 3 + 0];
    const float cy = p[last * 3 + 1];
    const float cz = p[last * 3 + 2];
    if (t == 0) {
      float* o = new_xyz + ((size_t)b * NPOINT + s) * 3;
      o[0] = cx; o[1] = cy; o[2] = cz;
    }
    float best = -1.f;
    int bi = 0;
#pragma unroll
    for (int i = 0; i < PPT; ++i) {
      const float dx = px[i] - cx, dy = py[i] - cy, dz = pz[i] - cz;
      const float d = dx * dx + dy * dy + dz * dz;
      dists[i] = fminf(dists[i], d);
      if (dists[i] > best) { best = dists[i]; bi = t + i * T; }
    }
    redv[t] = best; redi[t] = bi;
    __syncthreads();
    for (int off = T / 2; off > 0; off >>= 1) {
      if (t < off) {
        if (redv[t + off] > redv[t]) { redv[t] = redv[t + off]; redi[t] = redi[t + off]; }
      }
      __syncthreads();
    }
    if (t == 0) s_last = redi[0];
  }
}

// ---------------------------------------------------------------------------
// Ball query: one wave32 per (batch, center). In-order compaction of the
// first NS in-radius indices via a small LDS prefix scan; pad with the first
// in-radius index (torch ball_query semantics).
// ---------------------------------------------------------------------------
template <int N, int NS>
__global__ void __launch_bounds__(32) ball_query_kernel(
    const float* __restrict__ pts, const float* __restrict__ centers,
    int* __restrict__ nidx, float r2, int S) {
  __shared__ int pred[32];
  const int g = blockIdx.x;
  const int b = g / S, s = g - b * S;
  const int lane = threadIdx.x;
  const float* p = pts + (size_t)b * N * 3;
  const float* c = centers + ((size_t)b * S + s) * 3;
  const float cx = c[0], cy = c[1], cz = c[2];
  int* out = nidx + ((size_t)b * S + s) * NS;

  int count = 0;
  int firstIdx = -1;
  for (int base = 0; base < N; base += 32) {
    if (count >= NS) break;                       // uniform across the wave
    const int pi = base + lane;
    const float dx = p[pi * 3 + 0] - cx;
    const float dy = p[pi * 3 + 1] - cy;
    const float dz = p[pi * 3 + 2] - cz;
    const bool in = (dx * dx + dy * dy + dz * dz) < r2;
    pred[lane] = in ? 1 : 0;
    __syncthreads();
    int below = 0, tot = 0, firstj = -1;
    for (int j = 0; j < 32; ++j) {
      const int pj = pred[j];
      tot += pj;
      if (j < lane) below += pj;
      if (firstj < 0 && pj) firstj = j;
    }
    if (in) {
      const int slot = count + below;
      if (slot < NS) out[slot] = pi;
    }
    if (firstIdx < 0 && firstj >= 0) firstIdx = base + firstj;
    count += tot;
    __syncthreads();
  }
  if (firstIdx < 0) firstIdx = 0;
  for (int k = count + lane; k < NS; k += 32) out[k] = firstIdx;
}

// ---------------------------------------------------------------------------
// One WMMA MLP layer:  actOut = relu( actIn(NSxKPAD) * Wfrag + bias ), f16 out.
// Loop order mt -> kt -> owned nts: each A-fragment is loaded from LDS once
// and reused for every N-tile the wave owns (acc register-resident).
// ---------------------------------------------------------------------------
template <int NS, int KPAD, int COUT, int BLOCK>
__device__ __forceinline__ void mlp_layer(const _Float16* actIn, _Float16* actOut,
                                          const _Float16* __restrict__ Wfrag,
                                          const float* __restrict__ bias, int tid) {
  constexpr int MT = NS / 16, KT = KPAD / 32, NT = COUT / 16, NW = BLOCK / 32;
  constexpr int NTW = (NT + NW - 1) / NW;         // N-tiles per wave
  constexpr bool EXACT = (NT % NW) == 0;          // no tail guard needed
  const int wave = wave_id(tid);                  // SGPR (scalar guards)
  const int lane = tid & 31;
  const int nl = lane & 15, hi = lane >> 4;
  for (int mt = 0; mt < MT; ++mt) {
    v8f acc[NTW];
#pragma unroll
    for (int i = 0; i < NTW; ++i) acc[i] = vzero8();
    for (int kt = 0; kt < KT; ++kt) {
      const v16h a = load_A_frag(actIn, KPAD, mt, kt, lane);
      if (kt + 1 < KT)
        __builtin_prefetch(
            (const void*)(Wfrag + (((size_t)(kt + 1) * NT + wave) * 32 + lane) * 16), 0, 0);
#pragma unroll
      for (int i = 0; i < NTW; ++i) {
        const int nt = wave + i * NW;
        if (EXACT || nt < NT) {                   // folds away / scalar branch
          v16h bf = *(const v16h*)(Wfrag + (((size_t)kt * NT + nt) * 32 + lane) * 16);
          acc[i] = wmma_f32_16x16x32_f16(a, bf, acc[i]);
        }
      }
    }
#pragma unroll
    for (int i = 0; i < NTW; ++i) {
      const int nt = wave + i * NW;
      if (EXACT || nt < NT) {
        const float bn = bias[nt * 16 + nl];
#pragma unroll
        for (int v = 0; v < 8; ++v) {
          const int m = mt * 16 + v + 8 * hi;     // C/D layout: hi half-lane = M+8
          actOut[m * COUT + nt * 16 + nl] = (_Float16)fmaxf(acc[i][v] + bn, 0.f);
        }
      }
    }
  }
}

// Final layer fused with max-pool over the NS rows.
// red = LDS scratch of COUT*2 floats; outF = global (COUT floats).
template <int NS, int KPAD, int COUT, int BLOCK>
__device__ __forceinline__ void mlp_layer_maxpool(const _Float16* actIn,
                                                  const _Float16* __restrict__ Wfrag,
                                                  const float* __restrict__ bias,
                                                  float* red, float* __restrict__ outF,
                                                  int tid) {
  constexpr int MT = NS / 16, KT = KPAD / 32, NT = COUT / 16, NW = BLOCK / 32;
  constexpr int NTW = (NT + NW - 1) / NW;
  constexpr bool EXACT = (NT % NW) == 0;
  const int wave = wave_id(tid);
  const int lane = tid & 31, nl = lane & 15;
  float colmax[NTW];
#pragma unroll
  for (int i = 0; i < NTW; ++i) colmax[i] = 0.f;  // relu output >= 0
  for (int mt = 0; mt < MT; ++mt) {
    v8f acc[NTW];
#pragma unroll
    for (int i = 0; i < NTW; ++i) acc[i] = vzero8();
    for (int kt = 0; kt < KT; ++kt) {
      const v16h a = load_A_frag(actIn, KPAD, mt, kt, lane);
#pragma unroll
      for (int i = 0; i < NTW; ++i) {
        const int nt = wave + i * NW;
        if (EXACT || nt < NT) {
          v16h bf = *(const v16h*)(Wfrag + (((size_t)kt * NT + nt) * 32 + lane) * 16);
          acc[i] = wmma_f32_16x16x32_f16(a, bf, acc[i]);
        }
      }
    }
#pragma unroll
    for (int i = 0; i < NTW; ++i) {
      const int nt = wave + i * NW;
      if (EXACT || nt < NT) {
        const float bn = bias[nt * 16 + nl];
#pragma unroll
        for (int v = 0; v < 8; ++v)
          colmax[i] = fmaxf(colmax[i], fmaxf(acc[i][v] + bn, 0.f));
      }
    }
  }
#pragma unroll
  for (int i = 0; i < NTW; ++i) {
    const int nt = wave + i * NW;
    if (EXACT || nt < NT) red[nt * 32 + lane] = colmax[i];
  }
  __syncthreads();
  for (int n = tid; n < COUT; n += BLOCK) {
    const int nt = n >> 4, l = n & 15;
    outF[n] = fmaxf(red[nt * 32 + l], red[nt * 32 + l + 16]);
  }
}

// ---------------------------------------------------------------------------
// Set-abstraction group + 3-layer MLP + max-pool. One workgroup per (b, s).
// GROUP_ALL: use all NS points, no centering (SA3).
// ---------------------------------------------------------------------------
template <int NS, int CFEAT, int K0, int C1, int C2, int C3, bool GROUP_ALL, int BLOCK>
__global__ void __launch_bounds__(BLOCK) sa_mlp_kernel(
    const float* __restrict__ xyz,     // (B, Npts, 3)
    const float* __restrict__ feats,   // (B, Npts, CFEAT) or null
    const float* __restrict__ centers, // (B, S, 3) or null
    const int* __restrict__ nidx,      // (B, S, NS) or null
    const _Float16* __restrict__ Wf0, const float* __restrict__ b0,
    const _Float16* __restrict__ Wf1, const float* __restrict__ b1,
    const _Float16* __restrict__ Wf2, const float* __restrict__ b2,
    float* __restrict__ outF,          // (B, S, C3)
    int S, int Npts) {
  extern __shared__ char dynsmem[];
  constexpr int BUF0 = NS * (K0 > C2 ? K0 : C2);
  _Float16* buf0 = (_Float16*)dynsmem;          // layer1 input / layer3 input
  _Float16* buf1 = buf0 + BUF0;                 // layer1 out == layer2 input

  const int g = blockIdx.x;
  const int b = g / S, s = g - b * S;
  const int tid = threadIdx.x;
  const float* pb = xyz + (size_t)b * Npts * 3;
  const float* fb = feats ? (feats + (size_t)b * Npts * CFEAT) : nullptr;

  float cx = 0.f, cy = 0.f, cz = 0.f;
  const int* ni = nullptr;
  if (!GROUP_ALL) {
    const float* c = centers + ((size_t)b * S + s) * 3;
    cx = c[0]; cy = c[1]; cz = c[2];
    ni = nidx + ((size_t)b * S + s) * NS;
  }

  // Gather (xyz - center | feats | zero-pad) -> f16 LDS tile (NS x K0)
  for (int e = tid; e < NS * K0; e += BLOCK) {
    const int row = e / K0, k = e - row * K0;
    const int pt = GROUP_ALL ? row : ni[row];
    float v = 0.f;
    if (k < 3) {
      const float ck = (k == 0) ? cx : ((k == 1) ? cy : cz);
      v = pb[(size_t)pt * 3 + k] - ck;
    } else if (CFEAT > 0 && k < 3 + CFEAT) {
      v = fb[(size_t)pt * CFEAT + (k - 3)];
    }
    buf0[row * K0 + k] = (_Float16)v;
  }
  __syncthreads();
  mlp_layer<NS, K0, C1, BLOCK>(buf0, buf1, Wf0, b0, tid);
  __syncthreads();
  mlp_layer<NS, C1, C2, BLOCK>(buf1, buf0, Wf1, b1, tid);
  __syncthreads();
  mlp_layer_maxpool<NS, C2, C3, BLOCK>(buf0, Wf2, b2, (float*)buf1,
                                       outF + ((size_t)b * S + s) * C3, tid);
}

// ---------------------------------------------------------------------------
// FC head: (16,1024) -> 512 (LN,leaky) -> 256 (LN,leaky) -> 64 (tanh).
// Single block, 8 waves; M-tile = 16 = batch. Same A-reuse structure.
// ---------------------------------------------------------------------------
template <int K, int N>
__device__ __forceinline__ void fc_gemm(const _Float16* act,
                                        const _Float16* __restrict__ Wfrag,
                                        const float* __restrict__ bias,
                                        float* raw, int tid) {
  constexpr int KT = K / 32, NT = N / 16, NW = 8;
  constexpr int NTW = (NT + NW - 1) / NW;
  constexpr bool EXACT = (NT % NW) == 0;
  const int wave = wave_id(tid);
  const int lane = tid & 31, nl = lane & 15, hi = lane >> 4;
  v8f acc[NTW];
#pragma unroll
  for (int i = 0; i < NTW; ++i) acc[i] = vzero8();
  for (int kt = 0; kt < KT; ++kt) {
    const v16h a = load_A_frag(act, K, 0, kt, lane);
#pragma unroll
    for (int i = 0; i < NTW; ++i) {
      const int nt = wave + i * NW;
      if (EXACT || nt < NT) {
        v16h bf = *(const v16h*)(Wfrag + (((size_t)kt * NT + nt) * 32 + lane) * 16);
        acc[i] = wmma_f32_16x16x32_f16(a, bf, acc[i]);
      }
    }
  }
#pragma unroll
  for (int i = 0; i < NTW; ++i) {
    const int nt = wave + i * NW;
    if (EXACT || nt < NT) {
      const float bn = bias[nt * 16 + nl];
#pragma unroll
      for (int v = 0; v < 8; ++v)
        raw[(v + 8 * hi) * N + nt * 16 + nl] = acc[i][v] + bn;
    }
  }
}

__device__ __forceinline__ void ln_leaky(const float* raw, const float* __restrict__ gamma,
                                         const float* __restrict__ beta,
                                         _Float16* actOut, int N, int tid) {
  if (tid < 16) {
    const float* r = raw + tid * N;
    float mean = 0.f;
    for (int j = 0; j < N; ++j) mean += r[j];
    mean /= (float)N;
    float var = 0.f;
    for (int j = 0; j < N; ++j) { const float d = r[j] - mean; var += d * d; }
    var /= (float)N;
    const float rs = rsqrtf(var + EPSF);
    for (int j = 0; j < N; ++j) {
      float y = (r[j] - mean) * rs * gamma[j] + beta[j];
      y = (y > 0.f) ? y : 0.2f * y;
      actOut[tid * N + j] = (_Float16)y;
    }
  }
}

__global__ void __launch_bounds__(256) fc_head_kernel(
    const float* __restrict__ feat,                         // (16,1024)
    const _Float16* __restrict__ Wf1, const float* __restrict__ b1,
    const float* __restrict__ g1, const float* __restrict__ be1,
    const _Float16* __restrict__ Wf2, const float* __restrict__ b2,
    const float* __restrict__ g2, const float* __restrict__ be2,
    const _Float16* __restrict__ Wf3, const float* __restrict__ b3,
    float* __restrict__ out) {                              // (16,64)
  extern __shared__ char dynsmem[];
  _Float16* act = (_Float16*)dynsmem;                       // 16x1024 f16
  float* raw = (float*)(dynsmem + (size_t)16 * 1024 * sizeof(_Float16)); // 16x512 f32
  const int tid = threadIdx.x;
  for (int e = tid; e < 16 * 1024; e += 256) act[e] = (_Float16)feat[e];
  __syncthreads();
  fc_gemm<1024, 512>(act, Wf1, b1, raw, tid);
  __syncthreads();
  ln_leaky(raw, g1, be1, act, 512, tid);
  __syncthreads();
  fc_gemm<512, 256>(act, Wf2, b2, raw, tid);
  __syncthreads();
  ln_leaky(raw, g2, be2, act, 256, tid);
  __syncthreads();
  fc_gemm<256, 64>(act, Wf3, b3, raw, tid);
  __syncthreads();
  for (int e = tid; e < 16 * 64; e += 256) out[e] = tanhf(raw[e]);
}

// ---------------------------------------------------------------------------
// Host side.
// Input order (setup_inputs dict order):
//   0: xyz (16,4096,3)
//   1..15 : sa1 L0..L2, each {W, gamma, beta, mean, var}
//   16..30: sa2 L0..L2
//   31..45: sa3 L0..L2
//   46,47: fc1 {W,b}   48,49: fc2   50,51: fc3
//   52,53: ln1 {gamma,beta}   54,55: ln2 {gamma,beta}
// ---------------------------------------------------------------------------
extern "C" void kernel_launch(void* const* d_in, const int* in_sizes, int n_in,
                              void* d_out, int out_size, void* d_ws, size_t ws_size,
                              hipStream_t stream) {
  (void)in_sizes; (void)n_in; (void)out_size; (void)ws_size;
  const float* xyz = (const float*)d_in[0];
  const float *W[9], *Ga[9], *Be[9], *Mu[9], *Va[9];
  for (int l = 0; l < 9; ++l) {
    W[l]  = (const float*)d_in[1 + 5 * l + 0];
    Ga[l] = (const float*)d_in[1 + 5 * l + 1];
    Be[l] = (const float*)d_in[1 + 5 * l + 2];
    Mu[l] = (const float*)d_in[1 + 5 * l + 3];
    Va[l] = (const float*)d_in[1 + 5 * l + 4];
  }
  const float* fcW[3] = {(const float*)d_in[46], (const float*)d_in[48], (const float*)d_in[50]};
  const float* fcB[3] = {(const float*)d_in[47], (const float*)d_in[49], (const float*)d_in[51]};
  const float* ln1g = (const float*)d_in[52];
  const float* ln1b = (const float*)d_in[53];
  const float* ln2g = (const float*)d_in[54];
  const float* ln2b = (const float*)d_in[55];

  // Workspace carve-out (256B aligned).
  char* wsp = (char*)d_ws;
  auto alloc = [&](size_t bytes) -> void* {
    void* p = (void*)wsp;
    wsp += (bytes + 255) & ~(size_t)255;
    return p;
  };

  static const int CIN[12] = {3, 64, 64, 131, 128, 128, 259, 256, 512, 1024, 512, 256};
  static const int KP[12]  = {32, 64, 64, 160, 128, 128, 288, 256, 512, 1024, 512, 256};
  static const int CO[12]  = {64, 64, 128, 128, 128, 256, 256, 512, 1024, 512, 256, 64};
  _Float16* wf[12];
  float* bs[12];
  for (int l = 0; l < 12; ++l) {
    wf[l] = (_Float16*)alloc((size_t)KP[l] * CO[l] * sizeof(_Float16));
    bs[l] = (float*)alloc((size_t)CO[l] * sizeof(float));
  }
  float* nx1   = (float*)alloc((size_t)16 * 512 * 3 * 4);
  int*   nidx1 = (int*)  alloc((size_t)16 * 512 * 32 * 4);
  float* f1    = (float*)alloc((size_t)16 * 512 * 128 * 4);
  float* nx2   = (float*)alloc((size_t)16 * 128 * 3 * 4);
  int*   nidx2 = (int*)  alloc((size_t)16 * 128 * 64 * 4);
  float* f2    = (float*)alloc((size_t)16 * 128 * 256 * 4);
  float* feat3 = (float*)alloc((size_t)16 * 1024 * 4);

  // Fold BN into f16 fragment weights (9 SA layers + 3 FC layers).
  for (int l = 0; l < 12; ++l) {
    const int total = KP[l] * CO[l];
    const float *w, *g = nullptr, *be = nullptr, *m = nullptr, *v = nullptr, *lb = nullptr;
    if (l < 9) { w = W[l]; g = Ga[l]; be = Be[l]; m = Mu[l]; v = Va[l]; }
    else       { w = fcW[l - 9]; lb = fcB[l - 9]; }
    fold_weights_kernel<<<(total + 255) / 256, 256, 0, stream>>>(
        w, g, be, m, v, lb, wf[l], bs[l], CIN[l], KP[l], CO[l]);
  }

  // SA1
  fps_kernel<4096, 512><<<16, 256, 0, stream>>>(xyz, nx1);
  ball_query_kernel<4096, 32><<<16 * 512, 32, 0, stream>>>(xyz, nx1, nidx1, 0.2f * 0.2f, 512);
  {
    const size_t smem = ((size_t)32 * 64 + (size_t)32 * 64) * sizeof(_Float16);
    sa_mlp_kernel<32, 0, 32, 64, 64, 128, false, 128><<<16 * 512, 128, smem, stream>>>(
        xyz, nullptr, nx1, nidx1, wf[0], bs[0], wf[1], bs[1], wf[2], bs[2], f1, 512, 4096);
  }
  // SA2
  fps_kernel<512, 128><<<16, 256, 0, stream>>>(nx1, nx2);
  ball_query_kernel<512, 64><<<16 * 128, 32, 0, stream>>>(nx1, nx2, nidx2, 0.4f * 0.4f, 128);
  {
    const size_t smem = ((size_t)64 * 160 + (size_t)64 * 128) * sizeof(_Float16);
    sa_mlp_kernel<64, 128, 160, 128, 128, 256, false, 256><<<16 * 128, 256, smem, stream>>>(
        nx1, f1, nx2, nidx2, wf[3], bs[3], wf[4], bs[4], wf[5], bs[5], f2, 128, 512);
  }
  // SA3 (group-all)
  {
    const size_t smem = ((size_t)128 * 512 + (size_t)128 * 256) * sizeof(_Float16);
    sa_mlp_kernel<128, 256, 288, 256, 512, 1024, true, 256><<<16, 256, smem, stream>>>(
        nx2, f2, nullptr, nullptr, wf[6], bs[6], wf[7], bs[7], wf[8], bs[8], feat3, 1, 128);
  }
  // FC head
  {
    const size_t smem = (size_t)16 * 1024 * sizeof(_Float16) + (size_t)16 * 512 * sizeof(float);
    fc_head_kernel<<<1, 256, smem, stream>>>(feat3, wf[9], bs[9], ln1g, ln1b,
                                             wf[10], bs[10], ln2g, ln2b,
                                             wf[11], bs[11], (float*)d_out);
  }
}
